// TimeLSTM_28819230556835
// MI455X (gfx1250) — matrix operand order repaired and checked by reading, not compiled, not run on hardware
//
#include <hip/hip_runtime.h>

// TimeLSTM fused scan for MI455X (gfx1250, wave32, WMMA bf16).
// B=128, S=512, IN=H=512. 8 workgroups x 512 threads (16 waves), each WG owns
// a 16-row batch tile and runs the whole 512-step scan self-contained.
// Weights stay in L2 (4.5 MB bf16). Weight pointers are explicit global
// (address_space(1)) so loads lower to global_load_b128 saddr-form; a
// zero-valued but opaque SGPR offset keeps them inside the step loop
// (prevents LICM hoist + scratch spill).

static constexpr int kB  = 128;
static constexpr int kS  = 512;
static constexpr int kIN = 512;
static constexpr int kH  = 512;
static constexpr int kG4H = 4 * kH;   // 2048
static constexpr int kBT = 16;        // batch tile per workgroup
static constexpr int kWAVES = 16;     // waves per workgroup
static constexpr int kCOLS = kH / kWAVES; // 32 hidden cols per wave

typedef __attribute__((ext_vector_type(16))) __bf16 v16bf;
typedef __attribute__((ext_vector_type(8)))  float  v8f;
typedef __attribute__((ext_vector_type(4)))  unsigned int v4u;  // 128-bit chunk

typedef const __attribute__((address_space(1))) __bf16* gbf16p;
typedef const __attribute__((address_space(1))) v4u*    gv4up;

union FragBF {
  v4u   q[2];
  v16bf v;
};

// A-operand (16x32 bf16, MxK) from LDS row-major [16][512].
// lane 0-15: M=lane, K-half 0; lane 16-31: M=lane-16, K-half 16.
// VGPRs 0-3 hold K = k0+kh .. +7 ; VGPRs 4-7 hold K = k0+kh+8 .. +7
__device__ __forceinline__ v16bf load_a_lds(const __bf16* base, int lane, int k0) {
  const int m  = lane & 15;
  const int kh = (lane >> 4) << 4;          // 0 or 16
  const __bf16* p = base + m * 512 + k0 + kh;
  FragBF f;
  f.q[0] = *reinterpret_cast<const v4u*>(p);
  f.q[1] = *reinterpret_cast<const v4u*>(p + 8);
  return f.v;
}

// B-operand (32x16 bf16, KxN) for gates = act @ W^T  =>  B[k][n] = W[col0+n][k].
// lane holds column N = lane&15, 16 consecutive K (half selected by lane>=16).
// W is row-major [rows][512] bf16, in explicit global address space.
__device__ __forceinline__ v16bf load_b_glb(gbf16p w, int colbase, int lane, int k0) {
  const int n  = lane & 15;
  const int kh = (lane >> 4) << 4;          // 0 or 16
  gbf16p p = w + (size_t)(colbase + n) * 512 + k0 + kh;
  FragBF f;
  f.q[0] = *reinterpret_cast<gv4up>(p);
  f.q[1] = *reinterpret_cast<gv4up>(p + 8);
  return f.v;
}

__device__ __forceinline__ v8f wmma_bf16(v16bf a, v16bf b, v8f c) {
  return __builtin_amdgcn_wmma_f32_16x16x32_bf16(
      /*neg_a=*/false, a, /*neg_b=*/false, b,
      /*c_mod=*/(short)0, c, /*reuse_a=*/false, /*reuse_b=*/false);
}

__device__ __forceinline__ float fsigmoid(float x) {
  return 1.0f / (1.0f + __expf(-x));
}
__device__ __forceinline__ float ftanh(float x) {
  return 2.0f / (1.0f + __expf(-2.0f * x)) - 1.0f;
}

__global__ void convert_f32_to_bf16(const float* __restrict__ src,
                                    __bf16* __restrict__ dst, int n) {
  int i = blockIdx.x * blockDim.x + threadIdx.x;
  if (i < n) dst[i] = (__bf16)src[i];
}

__global__ void __launch_bounds__(512, 1)
time_lstm_scan(const float* __restrict__ inputs,      // [B,S,IN] f32
               const float* __restrict__ timestamps,  // [B,S]    f32
               const float* __restrict__ h0,          // [B,H]
               const float* __restrict__ c0,          // [B,H]
               const float* __restrict__ Wall_bias,   // [4H]
               const float* __restrict__ Uall_bias,   // [4H]
               const float* __restrict__ Wd_bias,     // [H]
               const __bf16* __restrict__ Wall,       // [4H][H]  bf16
               const __bf16* __restrict__ Uall,       // [4H][IN] bf16
               const __bf16* __restrict__ Wd,         // [H][H]   bf16
               float* __restrict__ out)               // outputs[B,S,H] | h[B,H] | c[B,H]
{
  __shared__ __attribute__((aligned(16))) __bf16 h_s[kBT * kH];
  __shared__ __attribute__((aligned(16))) __bf16 c_s[kBT * kH];
  __shared__ __attribute__((aligned(16))) __bf16 x_s[kBT * kIN];
  __shared__ float t_s[kBT];

  const int tid  = threadIdx.x;
  const int lane = tid & 31;
  const int wv   = tid >> 5;            // 0..15
  const int b0   = blockIdx.x * kBT;    // batch tile start
  const int col0 = wv * kCOLS;          // this wave's hidden columns

  const int nlo = lane & 15;            // N within a 16-wide tile
  const int mhi = (lane >> 4) << 3;     // +0 / +8 row offset in C layout

  // Explicit global-address-space weight pointers (saddr-form global loads).
  gbf16p wall_g = (gbf16p)Wall;
  gbf16p uall_g = (gbf16p)Uall;
  gbf16p wd_g   = (gbf16p)Wd;

  // ---- init LDS state from h0 / c0 ----
  for (int i = tid; i < kBT * kH; i += 512) {
    int r = i >> 9, c = i & (kH - 1);
    h_s[i] = (__bf16)h0[(size_t)(b0 + r) * kH + c];
    c_s[i] = (__bf16)c0[(size_t)(b0 + r) * kH + c];
  }

  // ---- c state in registers, WMMA C layout: creg[tile][r] = c[r+mhi][col0+tile*16+nlo]
  v8f creg[2];
#pragma unroll
  for (int t = 0; t < 2; ++t)
#pragma unroll
    for (int r = 0; r < 8; ++r)
      creg[t][r] = c0[(size_t)(b0 + r + mhi) * kH + col0 + t * 16 + nlo];

  // ---- per-lane bias constants (combined W_all_b + U_all_b; per-column) ----
  float bias_g[4][2];
#pragma unroll
  for (int g = 0; g < 4; ++g)
#pragma unroll
    for (int t = 0; t < 2; ++t) {
      int col = g * kH + col0 + t * 16 + nlo;
      bias_g[g][t] = Wall_bias[col] + Uall_bias[col];
    }
  float bias_d[2];
#pragma unroll
  for (int t = 0; t < 2; ++t) bias_d[t] = Wd_bias[col0 + t * 16 + nlo];

  const v8f vzero = {0.f, 0.f, 0.f, 0.f, 0.f, 0.f, 0.f, 0.f};

  // Opaque (always zero) uniform element offset: keeps every weight load
  // inside the step loop without obscuring the global base pointer.
  size_t woff = 0;

  __syncthreads();

  for (int s = 0; s < kS; ++s) {
    // Optimization barrier: compiler must assume woff changes each step.
    asm volatile("" : "+s"(woff));
    gbf16p wall_p = wall_g + woff;
    gbf16p uall_p = uall_g + woff;
    gbf16p wd_p   = wd_g + woff;

    // ---- phase 1: stage x_s (f32 -> bf16) and timestamps into LDS ----
    for (int i = tid; i < kBT * kIN; i += 512) {
      int r = i >> 9, c = i & (kIN - 1);
      x_s[i] = (__bf16)inputs[((size_t)(b0 + r) * kS + s) * kIN + c];
    }
    if (tid < kBT) t_s[tid] = timestamps[(size_t)(b0 + tid) * kS + s];
    __syncthreads();

    // ---- phase 2a: decay GEMM  c_s1_pre = c @ Wd^T ----
    v8f accd[2] = {vzero, vzero};
    for (int k = 0; k < kH; k += 32) {
      v16bf a = load_a_lds(c_s, lane, k);
      accd[0] = wmma_bf16(a, load_b_glb(wd_p, col0,      lane, k), accd[0]);
      accd[1] = wmma_bf16(a, load_b_glb(wd_p, col0 + 16, lane, k), accd[1]);
    }

    // ---- phase 2b: gates GEMM  gates_pre = h @ Wall^T + x @ Uall^T ----
    v8f acc[4][2];
#pragma unroll
    for (int g = 0; g < 4; ++g)
#pragma unroll
      for (int t = 0; t < 2; ++t) acc[g][t] = vzero;

    for (int k = 0; k < kH; k += 32) {
      v16bf ah = load_a_lds(h_s, lane, k);
      v16bf ax = load_a_lds(x_s, lane, k);
#pragma unroll
      for (int g = 0; g < 4; ++g)
#pragma unroll
        for (int t = 0; t < 2; ++t) {
          const int cb = g * kH + col0 + t * 16;
          acc[g][t] = wmma_bf16(ah, load_b_glb(wall_p, cb, lane, k), acc[g][t]);
          acc[g][t] = wmma_bf16(ax, load_b_glb(uall_p, cb, lane, k), acc[g][t]);
        }
    }
    __syncthreads();   // GEMM reads of h_s/c_s/x_s done before we overwrite

    // ---- phase 3: elementwise gate math + state update (all in registers) ----
    float tv[8];
#pragma unroll
    for (int r = 0; r < 8; ++r) tv[r] = t_s[r + mhi];

#pragma unroll
    for (int t = 0; t < 2; ++t) {
      const int col = col0 + t * 16 + nlo;
#pragma unroll
      for (int r = 0; r < 8; ++r) {
        const float cs1  = ftanh(accd[t][r] + bias_d[t]);
        const float cadj = (creg[t][r] - cs1) + cs1 * tv[r];
        const float fg = fsigmoid(acc[0][t][r] + bias_g[0][t]);
        const float ig = fsigmoid(acc[1][t][r] + bias_g[1][t]);
        const float og = fsigmoid(acc[2][t][r] + bias_g[2][t]);
        const float cg = fsigmoid(acc[3][t][r] + bias_g[3][t]);
        const float cn = fg * cadj + ig * cg;
        const float hn = og * ftanh(cn);
        creg[t][r] = cn;

        const int row = r + mhi;
        c_s[row * kH + col] = (__bf16)cn;
        h_s[row * kH + col] = (__bf16)hn;
        out[((size_t)(b0 + row) * kS + s) * kH + col] = og;
        if (s == kS - 1) {
          const size_t hc = (size_t)kB * kS * kH;
          out[hc + (size_t)(b0 + row) * kH + col] = hn;
          out[hc + (size_t)kB * kH + (size_t)(b0 + row) * kH + col] = cn;
        }
      }
    }
    __syncthreads();   // state writes visible before next step's GEMMs
  }
}

extern "C" void kernel_launch(void* const* d_in, const int* in_sizes, int n_in,
                              void* d_out, int out_size, void* d_ws, size_t ws_size,
                              hipStream_t stream) {
  const float* inputs     = (const float*)d_in[0];
  const float* timestamps = (const float*)d_in[1];
  const float* h0         = (const float*)d_in[2];
  const float* c0         = (const float*)d_in[3];
  const float* Wall_w     = (const float*)d_in[4];
  const float* Wall_b     = (const float*)d_in[5];
  const float* Uall_w     = (const float*)d_in[6];
  const float* Uall_b     = (const float*)d_in[7];
  const float* Wd_w       = (const float*)d_in[8];
  const float* Wd_b       = (const float*)d_in[9];
  float* out = (float*)d_out;

  // bf16 weight staging in workspace (4.5 MB total)
  __bf16* wall_bf = (__bf16*)d_ws;
  __bf16* uall_bf = wall_bf + (size_t)kG4H * kH;
  __bf16* wd_bf   = uall_bf + (size_t)kG4H * kIN;

  const int nW = kG4H * kH;   // 1,048,576
  const int nU = kG4H * kIN;  // 1,048,576
  const int nD = kH * kH;     //   262,144
  convert_f32_to_bf16<<<(nW + 255) / 256, 256, 0, stream>>>(Wall_w, wall_bf, nW);
  convert_f32_to_bf16<<<(nU + 255) / 256, 256, 0, stream>>>(Uall_w, uall_bf, nU);
  convert_f32_to_bf16<<<(nD + 255) / 256, 256, 0, stream>>>(Wd_w, wd_bf, nD);

  time_lstm_scan<<<kB / kBT, kWAVES * 32, 0, stream>>>(
      inputs, timestamps, h0, c0, Wall_b, Uall_b, Wd_b,
      wall_bf, uall_bf, wd_bf, out);
}
